// KMeansLayer_73023033967115
// MI455X (gfx1250) — matrix operand order repaired and checked by reading, not compiled, not run on hardware
//
#include <hip/hip_runtime.h>
#include <hip/hip_bf16.h>

// CDNA5 / gfx1250: wave32, WMMA 16x16x32 bf16 path.
typedef __attribute__((ext_vector_type(16))) __bf16 v16bf;
typedef __attribute__((ext_vector_type(8)))  float  v8f;

#define NPTS 65536
#define KCL  1024
#define DIM  256

// ---------------------------------------------------------------------------
// Kernel 1: split clusters (f32) into bf16 hi/lo, pre-swizzled into the
// WMMA B-fragment memory order so the GEMM kernel's B loads are 1KB
// contiguous blocks (32 lanes x 32B). Also compute exact f32 ||c||^2.
//
// B fragment layout (16x16x32 bf16, B is K(32) x N(16)):
//   lane n      (0..15) holds cluster (ct*16+n), K = kt*32 + {0..7, 16..23}
//   lane n+16           holds cluster (ct*16+n), K = kt*32 + {8..15, 24..31}
// linear: pos = ((ct*8 + kt)*32 + lane)*16 + j
// ---------------------------------------------------------------------------
__global__ __launch_bounds__(256) void prep_clusters_kernel(
    const float* __restrict__ clusters,
    __bf16* __restrict__ bhi, __bf16* __restrict__ blo,
    float* __restrict__ c2) {
  const int k = blockIdx.x;    // cluster id, 1024 blocks
  const int t = threadIdx.x;   // dim, 256 threads
  float x = clusters[k * DIM + t];
  __bf16 h = (__bf16)x;                       // hi (RNE truncation)
  __bf16 l = (__bf16)(x - (float)h);          // lo residual

  const int ct = k >> 4, n = k & 15;
  const int kt = t >> 5, r = t & 31;
  const int half = (r >> 3) & 1, j = (r & 7) | ((r >> 4) << 3);
  const int lane = half * 16 + n;
  const int pos = ((ct * 8 + kt) * 32 + lane) * 16 + j;
  bhi[pos] = h;
  blo[pos] = l;

  __shared__ float red[256];
  red[t] = x * x;
  __syncthreads();
  for (int s = 128; s > 0; s >>= 1) {
    if (t < s) red[t] += red[t + s];
    __syncthreads();
  }
  if (t == 0) c2[k] = red[0];
}

// ---------------------------------------------------------------------------
// Kernel 2: per block = 128 rows (8 waves x 16 rows). Each wave sweeps all
// 1024 clusters: 64 column tiles x 8 depth chunks x 3 bf16 WMMAs (hi*hi +
// hi*lo + lo*hi) with f32 accumulation. Track running argmin of
// ||c||^2 - 2 x.c (the ||x||^2 term is row-constant, irrelevant to argmin).
// Then gather the winning f32 centroid rows to the output (coalesced).
// ---------------------------------------------------------------------------
__global__ __launch_bounds__(256) void kmeans_assign_kernel(
    const float* __restrict__ inputs, const float* __restrict__ clusters,
    const __bf16* __restrict__ bhi, const __bf16* __restrict__ blo,
    const float* __restrict__ c2, float* __restrict__ out) {
  const int tid  = threadIdx.x;
  const int wave = tid >> 5;
  const int lane = tid & 31;
  const int m    = lane & 15;     // row within wave tile
  const int half = lane >> 4;     // A/B half-wave select
  const int row  = blockIdx.x * 128 + wave * 16 + m;

  // --- Load 16x256 A tile as f32, split to bf16 hi/lo in ISA A layout ---
  // lane 0-15 :  K = kt*32 + {0..7, 16..23}
  // lane 16-31:  K = kt*32 + {8..15, 24..31}
  v16bf ahi[8], alo[8];
  const float* __restrict__ arow = inputs + (size_t)row * DIM;
#pragma unroll
  for (int kt = 0; kt < 8; ++kt) {
    const int d0 = kt * 32 + half * 8;
    const float4* p0 = (const float4*)(arow + d0);        // K d0..d0+7
    const float4* p1 = (const float4*)(arow + d0 + 16);   // K d0+16..d0+23
    float4 f0 = p0[0], f1 = p0[1], f2 = p1[0], f3 = p1[1];
    float v[16] = {f0.x, f0.y, f0.z, f0.w, f1.x, f1.y, f1.z, f1.w,
                   f2.x, f2.y, f2.z, f2.w, f3.x, f3.y, f3.z, f3.w};
#pragma unroll
    for (int e = 0; e < 16; ++e) {
      __bf16 h = (__bf16)v[e];
      ahi[kt][e] = h;
      alo[kt][e] = (__bf16)(v[e] - (float)h);
    }
  }

  float minv[8];
  int   mini[8];
#pragma unroll
  for (int i = 0; i < 8; ++i) { minv[i] = 3.4e38f; mini[i] = 0; }

  // --- Sweep all 64 cluster tiles ---
  for (int ct = 0; ct < 64; ++ct) {
    v8f acc = {};
#pragma unroll
    for (int kt = 0; kt < 8; ++kt) {
      const size_t off = ((size_t)(ct * 8 + kt) * 32 + lane) * 16;
      v16bf bh = *(const v16bf*)(bhi + off);
      v16bf bl = *(const v16bf*)(blo + off);
      acc = __builtin_amdgcn_wmma_f32_16x16x32_bf16(
          false, ahi[kt], false, bh, (short)0, acc, false, false);
      acc = __builtin_amdgcn_wmma_f32_16x16x32_bf16(
          false, ahi[kt], false, bl, (short)0, acc, false, false);
      acc = __builtin_amdgcn_wmma_f32_16x16x32_bf16(
          false, alo[kt], false, bh, (short)0, acc, false, false);
    }
    // C layout: VGPR i, lanes 0-15 -> (M=i, N=lane); lanes 16-31 -> (M=8+i).
    // Cluster id for this lane is uniform across the 8 C registers.
    const int cid = ct * 16 + m;
    const float c2v = c2[cid];
#pragma unroll
    for (int i = 0; i < 8; ++i) {
      float s = c2v - 2.0f * acc[i];
      if (s < minv[i]) { minv[i] = s; mini[i] = cid; }
    }
  }

  // --- Argmin reduction across the 16 lanes of each half-wave ---
#pragma unroll
  for (int i = 0; i < 8; ++i) {
#pragma unroll
    for (int off = 8; off >= 1; off >>= 1) {
      float ov = __shfl_xor(minv[i], off, 32);
      int   oi = __shfl_xor(mini[i], off, 32);
      if (ov < minv[i] || (ov == minv[i] && oi < mini[i])) {
        minv[i] = ov; mini[i] = oi;
      }
    }
  }

  __shared__ int s_idx[128];
  if (m == 0) {
#pragma unroll
    for (int i = 0; i < 8; ++i) s_idx[wave * 16 + half * 8 + i] = mini[i];
  }
  __syncthreads();

  // --- Gather winning centroids (f32, bit-exact vs reference) ---
  const size_t out_base = (size_t)blockIdx.x * 128 * DIM;
  for (int r = 0; r < 128; ++r) {
    const int a = s_idx[r];
    out[out_base + (size_t)r * DIM + tid] = clusters[(size_t)a * DIM + tid];
  }
}

// ---------------------------------------------------------------------------
extern "C" void kernel_launch(void* const* d_in, const int* in_sizes, int n_in,
                              void* d_out, int out_size, void* d_ws, size_t ws_size,
                              hipStream_t stream) {
  const float* inputs   = (const float*)d_in[0];
  const float* clusters = (const float*)d_in[1];
  char* ws = (char*)d_ws;
  __bf16* bhi = (__bf16*)ws;                                        // 512 KB
  __bf16* blo = (__bf16*)(ws + (size_t)KCL * DIM * sizeof(__bf16)); // 512 KB
  float*  c2  = (float*)(ws + (size_t)2 * KCL * DIM * sizeof(__bf16)); // 4 KB
  float*  out = (float*)d_out;

  prep_clusters_kernel<<<KCL, 256, 0, stream>>>(clusters, bhi, blo, c2);
  kmeans_assign_kernel<<<NPTS / 128, 256, 0, stream>>>(inputs, clusters,
                                                       bhi, blo, c2, out);
}
